// TFLevitAttentionSubsample_26379689132086
// MI455X (gfx1250) — matrix operand order repaired and checked by reading, not compiled, not run on hardware
//
#include <hip/hip_runtime.h>
#include <hip/hip_bf16.h>

typedef __bf16 bf16_t;
typedef __attribute__((ext_vector_type(16))) __bf16 v16bf;
typedef __attribute__((ext_vector_type(8)))  float  v8f;
typedef __attribute__((ext_vector_type(4)))  unsigned int v4u;
typedef __attribute__((ext_vector_type(8)))  int  v8i;
typedef __attribute__((ext_vector_type(4)))  int  v4i;

#define HEADS   8
#define KEY_DIM 16
#define CIN     256
#define DKV     640
#define DQ      128
#define DPROJ   512
#define COUT    384
#define NTOK    196     // 14*14
#define NTOKP   224     // padded key-token count (7 * 32)
#define NQ      49      // 7*7
#define NBATCH  256
#define ATT_SCALE 0.25f // 16^-0.5

// ---------------------------------------------------------------------------
// WMMA helpers (bf16 A/B, f32 accumulate) — layouts per CDNA5 ISA
// A 16x32: lane holds row (lane&15); elems 0..7 -> K=(lane>>4)*8+j, 8..15 -> +16
// B 32x16: lane holds col (lane&15); elems 0..7 -> K=(lane>>4)*8+j, 8..15 -> +16
// C: lane l, reg r -> row r+8*(l>>4), col l&15
// ---------------------------------------------------------------------------
static __device__ __forceinline__ v8f wmma_bf16(v16bf a, v16bf b, v8f c) {
  return __builtin_amdgcn_wmma_f32_16x16x32_bf16(
      false, a, false, b, (short)0, c, false, false);
}

static __device__ __forceinline__ v16bf a_frag_bf16(const bf16_t* p, int kb0) {
  v16bf f;
#pragma unroll
  for (int j = 0; j < 8; ++j) { f[j] = p[kb0 + j]; f[j + 8] = p[16 + kb0 + j]; }
  return f;
}

static __device__ __forceinline__ v16bf b_frag_bf16(const bf16_t* p, int kb0b) {
  v16bf f;
#pragma unroll
  for (int j = 0; j < 8; ++j) { f[j] = p[kb0b / 2 + j]; f[j + 8] = p[16 + kb0b / 2 + j]; }
  return f;
}

// ---------------------------------------------------------------------------
// Tensor Data Mover: 2D tile load Global -> LDS (D# per CDNA5 ISA).
// Issued by one wave; caller must s_wait_tensorcnt + barrier before LDS use.
// data_code: 0=1B 1=2B 2=4B. dims/strides in data_size units.
// ---------------------------------------------------------------------------
#if __has_builtin(__builtin_amdgcn_tensor_load_to_lds)
#define USE_TDM 1
static __device__ __forceinline__ void tdm_load_2d(
    unsigned lds_addr, const void* gsrc, unsigned data_code,
    unsigned tile_d0, unsigned tile_d1,
    unsigned tens_d0, unsigned tens_d1, unsigned d0_stride) {
  unsigned long long ga = (unsigned long long)(size_t)gsrc;
  v4u g0;
  g0[0] = 1u;                                            // count=1 (valid), no gather
  g0[1] = lds_addr;                                      // bits 63:32
  g0[2] = (unsigned)(ga & 0xFFFFFFFFull);                // global_addr[31:0]
  g0[3] = (unsigned)((ga >> 32) & 0x01FFFFFFull) | (2u << 30);  // addr[56:32], type=2
  v8i g1;
  g1[0] = (int)(data_code << 16);                        // wg_mask=0, data_size
  g1[1] = (int)((tens_d0 & 0xFFFFu) << 16);              // tensor_dim0[15:0]
  g1[2] = (int)(((tens_d0 >> 16) & 0xFFFFu) | ((tens_d1 & 0xFFFFu) << 16));
  g1[3] = (int)(((tens_d1 >> 16) & 0xFFFFu) | (tile_d0 << 16));  // tile_dim0
  g1[4] = (int)(tile_d1 & 0xFFFFu);                      // tile_dim1 (tile_dim2=0)
  g1[5] = (int)d0_stride;                                // tensor_dim0_stride[31:0]
  g1[6] = 0;                                             // stride hi / dim1_stride
  g1[7] = 0;
  v4i g2 = {0, 0, 0, 0};
  v4i g3 = {0, 0, 0, 0};
#if defined(__clang_major__) && __clang_major__ >= 23
  v8i gx = {0, 0, 0, 0, 0, 0, 0, 0};
  __builtin_amdgcn_tensor_load_to_lds(g0, g1, g2, g3, gx, 0);
#else
  __builtin_amdgcn_tensor_load_to_lds(g0, g1, g2, g3, 0);
#endif
}
#else
#define USE_TDM 0
#endif

// ---------------------------------------------------------------------------
// Kernel 0: weights -> bf16 K-contiguous (B-operand) + expand attention bias
// into padded [H,49,224] fp32 table (pad = -1e30 doubles as softmax mask).
// ---------------------------------------------------------------------------
__global__ __launch_bounds__(256) void prep_kernel(
    const float* __restrict__ Wkv, const float* __restrict__ Wq,
    const float* __restrict__ Wp, const float* __restrict__ ab,
    const int* __restrict__ bidx, int n_off,
    bf16_t* __restrict__ WkvT, bf16_t* __restrict__ WqT,
    bf16_t* __restrict__ WpT, float* __restrict__ biasFull) {
  int i = blockIdx.x * blockDim.x + threadIdx.x;
  if (i < DKV * CIN)   { int n = i / CIN,   k = i % CIN;   WkvT[i] = (bf16_t)Wkv[k * DKV + n]; }
  if (i < DQ * CIN)    { int n = i / CIN,   k = i % CIN;   WqT[i]  = (bf16_t)Wq[k * DQ + n]; }
  if (i < COUT * DPROJ){ int n = i / DPROJ, k = i % DPROJ; WpT[i]  = (bf16_t)Wp[k * COUT + n]; }
  if (i < HEADS * NQ * NTOKP) {
    int h = i / (NQ * NTOKP), r = i % (NQ * NTOKP);
    int qi = r / NTOKP, kt = r % NTOKP;
    biasFull[i] = (kt < NTOK) ? ab[h * n_off + bidx[qi * NTOK + kt]] : -1e30f;
  }
}

// ---------------------------------------------------------------------------
// Kernel 1: fused KV + Q GEMM. One workgroup (8 waves) per batch.
// X staged once to LDS as bf16; A-fragments built once per M-tile and reused
// across all N-tiles. Per N-tile: one 16-load clause, one wait, 8 WMMAs.
// Work items: 0..12 kv M-tiles (40 N-tiles), 13..16 q M-tiles (8 N-tiles).
// ---------------------------------------------------------------------------
__global__ __launch_bounds__(256) void kvq_gemm(
    const float* __restrict__ X,
    const bf16_t* __restrict__ WkvT, const bf16_t* __restrict__ WqT,
    const float* __restrict__ g_kv, const float* __restrict__ b_kv,
    const float* __restrict__ g_q,  const float* __restrict__ b_q,
    bf16_t* __restrict__ kbuf, bf16_t* __restrict__ vT,
    bf16_t* __restrict__ qbuf) {
  const int b    = blockIdx.x;
  const int lane = threadIdx.x & 31;
  const int wave = threadIdx.x >> 5;
  const int hi   = lane >> 4;
  const int ln   = lane & 15;
  const int kb0  = hi * 8;
  const int kb0b = hi * 16;
  const float bninv = rsqrtf(1.0f + 1e-5f);

  __shared__ __align__(16) bf16_t Xlds[NTOK][CIN];   // 100,352 B

  // cooperative fp32 -> bf16 staging of X[b]
  {
    const float* Xb = X + (size_t)b * NTOK * CIN;
    for (int i = threadIdx.x * 4; i < NTOK * CIN; i += (int)blockDim.x * 4) {
      const float4 v = *(const float4*)(Xb + i);
      bf16_t* d = &Xlds[0][0] + i;
      d[0] = (bf16_t)v.x; d[1] = (bf16_t)v.y; d[2] = (bf16_t)v.z; d[3] = (bf16_t)v.w;
    }
  }
  __syncthreads();

  for (int it = wave; it < 17; it += 8) {
    const bool isq = (it >= 13);
    const int  mt  = isq ? it - 13 : it;
    int arow;
    if (!isq) {
      arow = mt * 16 + ln; if (arow >= NTOK) arow = NTOK - 1;
    } else {
      int qi = mt * 16 + ln; if (qi >= NQ) qi = NQ - 1;
      arow = (qi / 7) * 28 + (qi % 7) * 2;               // stride-2 subsample
    }
    const bf16_t* xr = &Xlds[arow][0];
    v16bf af[CIN / 32];
#pragma unroll
    for (int ks = 0; ks < CIN / 32; ++ks) af[ks] = a_frag_bf16(xr + ks * 32, kb0);

    const int nTiles = isq ? (DQ / 16) : (DKV / 16);
    const bf16_t* WT = isq ? WqT : WkvT;
    for (int nt = 0; nt < nTiles; ++nt) {
      const int ncol = nt * 16 + ln;
      const bf16_t* bp = WT + (size_t)ncol * CIN;
      // whole-tile B panel: 16 b128 loads issued together, one wait
      v16bf bb[CIN / 32];
#pragma unroll
      for (int ks = 0; ks < CIN / 32; ++ks) bb[ks] = b_frag_bf16(bp + ks * 32, kb0b);
      v8f acc = {};
#pragma unroll
      for (int ks = 0; ks < CIN / 32; ++ks) acc = wmma_bf16(af[ks], bb[ks], acc);

      if (!isq) {
        const float gs = g_kv[ncol] * bninv, bs = b_kv[ncol];
        const int h = ncol / 80, dd = ncol % 80;
#pragma unroll
        for (int r = 0; r < 8; ++r) {
          const int orow = mt * 16 + r + (hi << 3);
          if (orow < NTOK) {
            const float v = acc[r] * gs + bs;
            if (dd < KEY_DIM)
              kbuf[(((size_t)b * HEADS + h) * NTOK + orow) * KEY_DIM + dd] = (bf16_t)v;
            else
              vT[(((size_t)b * HEADS + h) * 64 + (dd - KEY_DIM)) * NTOKP + orow] = (bf16_t)v;
          }
        }
      } else {
        const float gs = g_q[ncol] * bninv, bs = b_q[ncol];
        const int h = ncol >> 4, dd = ncol & 15;
#pragma unroll
        for (int r = 0; r < 8; ++r) {
          const int oq = mt * 16 + r + (hi << 3);
          if (oq < NQ)
            qbuf[(((size_t)b * HEADS + h) * NQ + oq) * KEY_DIM + dd] =
                (bf16_t)((acc[r] * gs + bs) * ATT_SCALE);
        }
      }
    }
  }
}

// ---------------------------------------------------------------------------
// Kernel 2: attention. One workgroup per (b,h); wave w owns q-row tile w.
// v^T slice staged to LDS via Tensor Data Mover; scores in registers;
// shfl-based softmax; P round-trips LDS to reshape into A layout for P@V.
// ---------------------------------------------------------------------------
__global__ __launch_bounds__(128) void attn_kernel(
    const bf16_t* __restrict__ qbuf, const bf16_t* __restrict__ kbuf,
    const bf16_t* __restrict__ vT, const float* __restrict__ biasFull,
    bf16_t* __restrict__ outb) {
  const int bh   = blockIdx.x;
  const int b    = bh / HEADS;
  const int h    = bh % HEADS;
  const int lane = threadIdx.x & 31;
  const int wave = threadIdx.x >> 5;
  const int hi   = lane >> 4;
  const int ln   = lane & 15;
  const int kb0  = hi * 8;
  const int kb0b = hi * 16;

  __shared__ __align__(16) bf16_t P[4][16][NTOKP];     // 28,672 B
  __shared__ __align__(16) bf16_t Vlds[64][NTOKP];     // 28,672 B

  const bf16_t* vbh = vT + (size_t)bh * 64 * NTOKP;
#if USE_TDM
  if (wave == 0) {
    tdm_load_2d((unsigned)(size_t)&Vlds[0][0], vbh, /*2B*/1,
                NTOKP, 64, NTOKP, 64, NTOKP);
    __builtin_amdgcn_s_wait_tensorcnt(0);
  }
#else
  {
    const uint4* s = (const uint4*)vbh;
    uint4* d = (uint4*)&Vlds[0][0];
    for (int i = threadIdx.x; i < 64 * NTOKP * 2 / 16; i += (int)blockDim.x) d[i] = s[i];
  }
#endif
  {
    bf16_t* pw = &P[wave][0][0];
    for (int i = lane; i < 16 * NTOKP; i += 32) pw[i] = (bf16_t)0.0f;
  }
  __syncthreads();

  // q A-fragment: K 0..15 valid, 16..31 zero-padded
  int qld = wave * 16 + ln; if (qld >= NQ) qld = NQ - 1;
  const bf16_t* qrow = qbuf + ((size_t)bh * NQ + qld) * KEY_DIM;
  v16bf aq;
#pragma unroll
  for (int j = 0; j < 8; ++j) { aq[j] = qrow[kb0 + j]; aq[j + 8] = (bf16_t)0.0f; }

  // ---- scores (13 key tiles, kept in registers) ----
  float sc[13][8];
  const bf16_t* kbh = kbuf + (size_t)bh * NTOK * KEY_DIM;
  const float* bsrc = biasFull + (size_t)h * NQ * NTOKP;
#pragma unroll
  for (int nt = 0; nt < 13; ++nt) {
    const int kt = nt * 16 + ln;
    const int ktc = (kt < NTOK) ? kt : NTOK - 1;
    const bf16_t* krow = kbh + (size_t)ktc * KEY_DIM;
    v16bf bk;
#pragma unroll
    for (int j = 0; j < 16; ++j) {
      const int kk = kb0 + j + ((j >> 3) << 3);
      bk[j] = (kk < KEY_DIM) ? krow[kk] : (bf16_t)0.0f;
    }
    v8f acc = {};
    acc = wmma_bf16(aq, bk, acc);
#pragma unroll
    for (int r = 0; r < 8; ++r) {
      const int qi = wave * 16 + r + (hi << 3);
      sc[nt][r] = (qi < NQ) ? (acc[r] + bsrc[qi * NTOKP + kt]) : -1e30f;
    }
  }

  // ---- softmax over keys (rows live in 16-lane halves) ----
#pragma unroll
  for (int r = 0; r < 8; ++r) {
    float mx = -1e30f;
#pragma unroll
    for (int nt = 0; nt < 13; ++nt) mx = fmaxf(mx, sc[nt][r]);
    mx = fmaxf(mx, __shfl_xor(mx, 1, 16));
    mx = fmaxf(mx, __shfl_xor(mx, 2, 16));
    mx = fmaxf(mx, __shfl_xor(mx, 4, 16));
    mx = fmaxf(mx, __shfl_xor(mx, 8, 16));
    float s = 0.0f;
#pragma unroll
    for (int nt = 0; nt < 13; ++nt) {
      const float e = __expf(sc[nt][r] - mx);
      sc[nt][r] = e;
      s += e;
    }
    s += __shfl_xor(s, 1, 16);
    s += __shfl_xor(s, 2, 16);
    s += __shfl_xor(s, 4, 16);
    s += __shfl_xor(s, 8, 16);
    const float inv = 1.0f / s;
    const int m = r + (hi << 3);
#pragma unroll
    for (int nt = 0; nt < 13; ++nt)
      P[wave][m][nt * 16 + ln] = (bf16_t)(sc[nt][r] * inv);
  }
  __syncthreads();

  // ---- out = P @ V : K = 224 (7 steps), N = 64; both operands in LDS ----
  v8f oacc[4] = {{}, {}, {}, {}};
#pragma unroll
  for (int ks = 0; ks < 7; ++ks) {
    const bf16_t* prow = &P[wave][ln][ks * 32];
    v16bf ap;
#pragma unroll
    for (int j = 0; j < 8; ++j) { ap[j] = prow[kb0 + j]; ap[j + 8] = prow[16 + kb0 + j]; }
#pragma unroll
    for (int nt2 = 0; nt2 < 4; ++nt2)
      oacc[nt2] = wmma_bf16(ap, b_frag_bf16(&Vlds[nt2 * 16 + ln][ks * 32], kb0b), oacc[nt2]);
  }

  // hardswish + store [B, 49, H*64] bf16
#pragma unroll
  for (int nt2 = 0; nt2 < 4; ++nt2) {
#pragma unroll
    for (int r = 0; r < 8; ++r) {
      const int qi = wave * 16 + r + (hi << 3);
      if (qi < NQ) {
        const float o  = oacc[nt2][r];
        const float hs = o * fminf(fmaxf(o + 3.0f, 0.0f), 6.0f) * (1.0f / 6.0f);
        outb[((size_t)b * NQ + qi) * DPROJ + h * 64 + nt2 * 16 + ln] = (bf16_t)hs;
      }
    }
  }
}

// ---------------------------------------------------------------------------
// Kernel 3: result = BN(out @ W_p) -> fp32. A slice staged to LDS via TDM;
// wave owns one M-tile, preloads all 16 A-fragments, streams B in two
// 8-fragment panels per tile (one wait per panel, 8 WMMAs back-to-back).
// ---------------------------------------------------------------------------
__global__ __launch_bounds__(128) void proj_gemm(
    const bf16_t* __restrict__ outb, const bf16_t* __restrict__ WpT,
    const float* __restrict__ g_p, const float* __restrict__ b_p,
    float* __restrict__ out) {
  const int b    = blockIdx.x;
  const int lane = threadIdx.x & 31;
  const int wave = threadIdx.x >> 5;   // = M-tile
  const int hi   = lane >> 4;
  const int ln   = lane & 15;
  const int kb0  = hi * 8;
  const int kb0b = hi * 16;
  const float bninv = rsqrtf(1.0f + 1e-5f);

  __shared__ __align__(16) bf16_t Olds[NQ][DPROJ];     // 50,176 B
  const bf16_t* osrc = outb + (size_t)b * NQ * DPROJ;
#if USE_TDM
  if (wave == 0) {
    tdm_load_2d((unsigned)(size_t)&Olds[0][0], osrc, /*2B*/1,
                DPROJ, NQ, DPROJ, NQ, DPROJ);
    __builtin_amdgcn_s_wait_tensorcnt(0);
  }
#else
  {
    const uint4* s = (const uint4*)osrc;
    uint4* d = (uint4*)&Olds[0][0];
    for (int i = threadIdx.x; i < NQ * DPROJ * 2 / 16; i += (int)blockDim.x) d[i] = s[i];
  }
#endif
  __syncthreads();

  int qi = wave * 16 + ln; if (qi >= NQ) qi = NQ - 1;
  const bf16_t* ar = &Olds[qi][0];
  v16bf af[DPROJ / 32];
#pragma unroll
  for (int ks = 0; ks < DPROJ / 32; ++ks) af[ks] = a_frag_bf16(ar + ks * 32, kb0);

  for (int nt = 0; nt < COUT / 16; ++nt) {
    const int ncol = nt * 16 + ln;
    const bf16_t* bp = WpT + (size_t)ncol * DPROJ;
    v8f acc = {};
#pragma unroll
    for (int half = 0; half < 2; ++half) {
      v16bf bb[8];
#pragma unroll
      for (int k2 = 0; k2 < 8; ++k2)
        bb[k2] = b_frag_bf16(bp + (half * 8 + k2) * 32, kb0b);
#pragma unroll
      for (int k2 = 0; k2 < 8; ++k2)
        acc = wmma_bf16(af[half * 8 + k2], bb[k2], acc);
    }
    const float gs = g_p[ncol] * bninv, bs = b_p[ncol];
#pragma unroll
    for (int r = 0; r < 8; ++r) {
      const int oq = wave * 16 + r + (hi << 3);
      if (oq < NQ)
        out[((size_t)b * NQ + oq) * COUT + ncol] = acc[r] * gs + bs;
    }
  }
}

// ---------------------------------------------------------------------------
extern "C" void kernel_launch(void* const* d_in, const int* in_sizes, int n_in,
                              void* d_out, int out_size, void* d_ws, size_t ws_size,
                              hipStream_t stream) {
  (void)n_in; (void)out_size; (void)ws_size;
  const float* X    = (const float*)d_in[0];
  const float* Wkv  = (const float*)d_in[1];
  const float* g_kv = (const float*)d_in[2];
  const float* b_kv = (const float*)d_in[3];
  const float* Wq   = (const float*)d_in[4];
  const float* g_q  = (const float*)d_in[5];
  const float* b_q  = (const float*)d_in[6];
  const float* Wp   = (const float*)d_in[7];
  const float* g_p  = (const float*)d_in[8];
  const float* b_p  = (const float*)d_in[9];
  const float* ab   = (const float*)d_in[10];
  const int*   bidx = (const int*)d_in[11];
  float* out = (float*)d_out;
  const int n_off = in_sizes[10] / HEADS;

  char* ws = (char*)d_ws;
  size_t off = 0;
  auto alloc = [&](size_t bytes) -> void* {
    void* p = ws + off;
    off += (bytes + 255) & ~(size_t)255;
    return p;
  };
  bf16_t* WkvT = (bf16_t*)alloc((size_t)DKV * CIN * sizeof(bf16_t));
  bf16_t* WqT  = (bf16_t*)alloc((size_t)DQ * CIN * sizeof(bf16_t));
  bf16_t* WpT  = (bf16_t*)alloc((size_t)COUT * DPROJ * sizeof(bf16_t));
  float*  biasFull = (float*)alloc((size_t)HEADS * NQ * NTOKP * sizeof(float));
  bf16_t* kbuf = (bf16_t*)alloc((size_t)NBATCH * HEADS * NTOK * KEY_DIM * sizeof(bf16_t));
  const size_t vT_bytes = (size_t)NBATCH * HEADS * 64 * NTOKP * sizeof(bf16_t);
  bf16_t* vT   = (bf16_t*)alloc(vT_bytes);
  bf16_t* qbuf = (bf16_t*)alloc((size_t)NBATCH * HEADS * NQ * KEY_DIM * sizeof(bf16_t));
  bf16_t* outb = (bf16_t*)alloc((size_t)NBATCH * NQ * DPROJ * sizeof(bf16_t));

  hipMemsetAsync(vT, 0, vT_bytes, stream);   // zero pad columns of v^T
  prep_kernel<<<768, 256, 0, stream>>>(Wkv, Wq, Wp, ab, bidx, n_off,
                                       WkvT, WqT, WpT, biasFull);
  kvq_gemm<<<NBATCH, 256, 0, stream>>>(X, WkvT, WqT, g_kv, b_kv, g_q, b_q,
                                       kbuf, vT, qbuf);
  attn_kernel<<<NBATCH * HEADS, 128, 0, stream>>>(qbuf, kbuf, vT, biasFull, outb);
  proj_gemm<<<NBATCH, 128, 0, stream>>>(outb, WpT, g_p, b_p, out);
}
